// T5GemmaAttention_35777077576348
// MI455X (gfx1250) — matrix operand order repaired and checked by reading, not compiled, not run on hardware
//
#include <hip/hip_runtime.h>
#include <hip/hip_bf16.h>

// ---------------------------------------------------------------------------
// T5Gemma attention block on gfx1250 (MI455X), bf16 WMMA with f32 accumulate.
//   B=2, S=2048, D=2048, H=16, Hkv=4, hd=128, softcap=clip(+-50), causal
// Reference quirks preserved:
//   * q is MULTIPLIED by sqrt(128)
//   * scores are clipped AFTER the -1e9 causal mask, so masked entries are
//     exactly -50 and contribute exp(-50 - m) to the softmax denominator.
// All operand fragments are loaded from LDS as contiguous 16-byte runs
// (B operands stored transposed [N][K]) so frag loads lower to ds_load_b128.
// ---------------------------------------------------------------------------

typedef __bf16 bf16_t;
typedef __attribute__((ext_vector_type(16))) __bf16 v16bf;
typedef __attribute__((ext_vector_type(8)))  float  v8f;

#define D_MODEL 2048
#define SEQ     2048
#define NHEADS  16
#define NKV     4
#define HDIM    128

__device__ __forceinline__ v8f wmma_bf16(v16bf a, v16bf b, v8f c) {
  return __builtin_amdgcn_wmma_f32_16x16x32_bf16(
      /*neg_a=*/false, a, /*neg_b=*/false, b,
      /*c_mod=*/(short)0, c, /*reuse_a=*/false, /*reuse_b=*/false);
}

// A-fragment (16x32, 16-bit, row-major [M][K] storage):
// lane L(0-15): row L, K = {kh..kh+7, 16+kh..16+kh+7}, kh = 8*(lane>>4).
// Two contiguous 16B runs per lane -> 2x ds_load_b128 (LD*2 multiple of 16).
template<int LD>
__device__ __forceinline__ v16bf load_frag_a(const bf16_t* base) {
  const int lane = threadIdx.x & 31;
  const bf16_t* p = base + (lane & 15) * LD + ((lane >> 4) << 3);
  v16bf r;
#pragma unroll
  for (int i = 0; i < 8; ++i) { r[i] = p[i]; r[8 + i] = p[16 + i]; }
  return r;
}

// B-fragment (32x16) from TRANSPOSED storage [N][K]:
// lane L(0-15): col L, K = {kh..kh+7, 16+kh..16+kh+7}, kh = 8*(lane>>4).
// Two contiguous 16B runs per lane -> 2x ds_load_b128.
template<int LD>
__device__ __forceinline__ v16bf load_frag_bt(const bf16_t* base) {
  const int lane = threadIdx.x & 31;
  const bf16_t* p = base + (lane & 15) * LD + ((lane >> 4) << 3);
  v16bf r;
#pragma unroll
  for (int i = 0; i < 8; ++i) { r[i] = p[i]; r[8 + i] = p[16 + i]; }
  return r;
}

// ---------------------------------------------------------------------------
// Kernel 0: tiled transpose + f32->bf16 convert:  src[K][N] -> dst[N][K].
// One-time pass over the weights; negligible vs compute (roofline: ~4us).
// ---------------------------------------------------------------------------
__global__ __launch_bounds__(256)
void transpose_cvt_kernel(const float* __restrict__ src, int K, int N,
                          bf16_t* __restrict__ dst) {
  __shared__ bf16_t tile[32][33];
  const int nb = blockIdx.x * 32, kb = blockIdx.y * 32;
  const int tx = threadIdx.x & 31, ty = threadIdx.x >> 5;
#pragma unroll
  for (int r = ty; r < 32; r += 8)
    tile[r][tx] = (bf16_t)src[(size_t)(kb + r) * N + nb + tx];
  __syncthreads();
#pragma unroll
  for (int r = ty; r < 32; r += 8)
    dst[(size_t)(nb + r) * K + kb + tx] = tile[tx][r];
}

// ---------------------------------------------------------------------------
// Kernel 1: fused QKV projection.  C tile 128x128, 8 waves (wave = 32x64),
// K-step 64.  WT = [Wq^T | Wk^T | Wv^T] bf16, rows = output column, ld = 2048.
// grid: x = n-tile (24 over 3072), y = m-tile (32 over 4096 rows)
// ---------------------------------------------------------------------------
__global__ __launch_bounds__(256)
void qkv_gemm_kernel(const float* __restrict__ X,
                     const bf16_t* __restrict__ WT,
                     bf16_t* __restrict__ Qbuf,   // [B][H][S][128], pre-scaled sqrt(128)
                     bf16_t* __restrict__ Kbuf,   // [B][Hkv][S][128]  (natural)
                     bf16_t* __restrict__ Vt)     // [B][Hkv][128][S]  (transposed)
{
  const int n0 = blockIdx.x * 128;
  const int m0 = blockIdx.y * 128;
  const int tid = threadIdx.x, wave = tid >> 5, lane = tid & 31;

  __shared__ __attribute__((aligned(16))) bf16_t As[128][72];  // [M][K] bf16
  __shared__ __attribute__((aligned(16))) bf16_t Bs[128][72];  // [N][K] bf16

  const int wm = wave & 3;   // 4 wave-rows x 32
  const int wn = wave >> 2;  // 2 wave-cols x 64

  v8f acc[2][4] = {};

  for (int k0 = 0; k0 < D_MODEL; k0 += 64) {
    for (int i = tid; i < 128 * 16; i += 256) {      // A: 128x64 f32 -> bf16
      int r = i >> 4, c4 = (i & 15) * 4;
      float4 f = *(const float4*)(X + (size_t)(m0 + r) * D_MODEL + k0 + c4);
      union { bf16_t h[4]; uint2 u; } t;
      t.h[0] = (bf16_t)f.x; t.h[1] = (bf16_t)f.y;
      t.h[2] = (bf16_t)f.z; t.h[3] = (bf16_t)f.w;
      *(uint2*)&As[r][c4] = t.u;
    }
    for (int i = tid; i < 128 * 16; i += 256) {      // B: 128 cols x 64 K (bf16 copy)
      int r = i >> 4, c4 = (i & 15) * 4;
      *(uint2*)&Bs[r][c4] = *(const uint2*)(WT + (size_t)(n0 + r) * D_MODEL + k0 + c4);
    }
    __syncthreads();

#pragma unroll
    for (int kc = 0; kc < 2; ++kc) {
      v16bf a0 = load_frag_a<72>(&As[wm * 32 + 0][kc * 32]);
      v16bf a1 = load_frag_a<72>(&As[wm * 32 + 16][kc * 32]);
#pragma unroll
      for (int nj = 0; nj < 4; ++nj) {
        v16bf bfr = load_frag_bt<72>(&Bs[wn * 64 + nj * 16][kc * 32]);
        acc[0][nj] = wmma_bf16(a0, bfr, acc[0][nj]);
        acc[1][nj] = wmma_bf16(a1, bfr, acc[1][nj]);
      }
    }
    __syncthreads();
  }

  const float qscale = 11.313708498984761f;  // sqrt(128): reference multiplies
#pragma unroll
  for (int mi = 0; mi < 2; ++mi)
#pragma unroll
    for (int nj = 0; nj < 4; ++nj)
#pragma unroll
      for (int i = 0; i < 8; ++i) {
        int ml = wm * 32 + mi * 16 + i + ((lane >> 4) << 3);
        int nl = wn * 64 + nj * 16 + (lane & 15);
        int M = m0 + ml, N = n0 + nl;
        int b = M >> 11, s = M & 2047;
        float v = acc[mi][nj][i];
        if (N < 2048) {
          int h = N >> 7, d = N & 127;
          Qbuf[(((size_t)(b * NHEADS + h) * SEQ + s) << 7) + d] = (bf16_t)(v * qscale);
        } else if (N < 2560) {
          int kh = (N - 2048) >> 7, d = N & 127;
          Kbuf[(((size_t)(b * NKV + kh) * SEQ + s) << 7) + d] = (bf16_t)v;
        } else {
          int kh = (N - 2560) >> 7, d = N & 127;
          Vt[(((size_t)(b * NKV + kh) * HDIM + d) << 11) + s] = (bf16_t)v;
        }
      }
}

// ---------------------------------------------------------------------------
// Kernel 2: flash-style causal attention with clip soft-cap.
// block = (q-tile of 128 rows, head, batch); 8 waves, wave owns 16 q-rows.
// Masked entries are exactly -50 (clip(score-1e9)), so all 16 key tiles run.
// LDS (dynamic, 102 KB; WGP has 320 KB):
//   Ks  [128 key][136 d]  (natural K  -> B-frags for Q@K^T contiguous)
//   Vts [128 d ][136 key] (V^T        -> B-frags for P@V   contiguous)
//   Ps  [8 waves][16][136]
// ---------------------------------------------------------------------------
__global__ __launch_bounds__(256)
void flash_attn_kernel(const bf16_t* __restrict__ Qbuf,
                       const bf16_t* __restrict__ Kbuf,
                       const bf16_t* __restrict__ Vt,
                       bf16_t* __restrict__ CtxBuf)  // [B][S][H*128]
{
  extern __shared__ __attribute__((aligned(16))) char smem[];
  const int LKV = 136;
  bf16_t* Ks  = (bf16_t*)smem;            // [128][136]
  bf16_t* Vts = Ks + 128 * LKV;           // [128][136]
  bf16_t* Ps  = Vts + 128 * LKV;          // [8][16][136]

  const int qt = blockIdx.x, h = blockIdx.y, b = blockIdx.z;
  const int kvh = h >> 2;                 // GQA: 4 query heads per kv head
  const int tid = threadIdx.x, wave = tid >> 5, lane = tid & 31;
  const int q0 = qt * 128;
  bf16_t* Pw = Ps + wave * 16 * LKV;

  const bf16_t* Qh  = Qbuf + ((size_t)(b * NHEADS + h) * SEQ) * HDIM;
  const bf16_t* Kh  = Kbuf + ((size_t)(b * NKV + kvh) * SEQ) * HDIM;
  const bf16_t* Vth = Vt   + ((size_t)(b * NKV + kvh) * HDIM) * SEQ;

  // preload Q A-fragments for this wave's 16 rows (already scaled)
  v16bf qf[4];
#pragma unroll
  for (int kc = 0; kc < 4; ++kc)
    qf[kc] = load_frag_a<HDIM>(Qh + (size_t)(q0 + wave * 16) * HDIM + kc * 32);

  v8f ctx[8] = {};
  float rm[8], rl[8];
#pragma unroll
  for (int i = 0; i < 8; ++i) { rm[i] = -1e30f; rl[i] = 0.f; }
  const int qrow_base = q0 + wave * 16 + ((lane >> 4) << 3);  // row for acc VGPR i = base+i
  const int ncol = lane & 15;

  for (int kt = 0; kt < SEQ / 128; ++kt) {
    const int kbase = kt * 128;
    for (int i = tid; i < 128 * 32; i += 256) {   // Ks <- K[kbase+r][d]
      int r = i >> 5, c4 = (i & 31) * 4;
      *(uint2*)&Ks[r * LKV + c4] = *(const uint2*)(Kh + (size_t)(kbase + r) * HDIM + c4);
    }
    for (int i = tid; i < 128 * 32; i += 256) {   // Vts <- Vt[d][kbase+c]
      int r = i >> 5, c4 = (i & 31) * 4;
      *(uint2*)&Vts[r * LKV + c4] = *(const uint2*)(Vth + (size_t)r * SEQ + kbase + c4);
    }
    __syncthreads();

    // S = Q @ K^T  (16 x 128 per wave); B-frag cols = keys, K-dim = head dim
    v8f sacc[8] = {};
#pragma unroll
    for (int kc = 0; kc < 4; ++kc)
#pragma unroll
      for (int nj = 0; nj < 8; ++nj) {
        v16bf bfr = load_frag_bt<136>(Ks + (nj * 16) * LKV + kc * 32);
        sacc[nj] = wmma_bf16(qf[kc], bfr, sacc[nj]);
      }

    // clip soft-cap + causal (masked -> exactly -50), tile row-max
    float tmax[8];
#pragma unroll
    for (int i = 0; i < 8; ++i) tmax[i] = -1e30f;
#pragma unroll
    for (int nj = 0; nj < 8; ++nj) {
      int key = kbase + nj * 16 + ncol;
#pragma unroll
      for (int i = 0; i < 8; ++i) {
        float s = fminf(50.f, fmaxf(-50.f, sacc[nj][i]));
        if (key > qrow_base + i) s = -50.f;
        sacc[nj][i] = s;
        tmax[i] = fmaxf(tmax[i], s);
      }
    }
#pragma unroll
    for (int i = 0; i < 8; ++i) {   // reduce over the 16 lanes sharing each row
      float m = tmax[i];
      m = fmaxf(m, __shfl_xor(m, 1, 32));
      m = fmaxf(m, __shfl_xor(m, 2, 32));
      m = fmaxf(m, __shfl_xor(m, 4, 32));
      m = fmaxf(m, __shfl_xor(m, 8, 32));
      tmax[i] = m;
    }
    float corr[8], tsum[8];
#pragma unroll
    for (int i = 0; i < 8; ++i) {
      float nm = fmaxf(rm[i], tmax[i]);
      corr[i] = __expf(rm[i] - nm);
      rm[i] = nm;
      tsum[i] = 0.f;
    }
    // exp, stash P (C-layout -> LDS -> A-layout), partial sums
#pragma unroll
    for (int nj = 0; nj < 8; ++nj)
#pragma unroll
      for (int i = 0; i < 8; ++i) {
        float p = __expf(sacc[nj][i] - rm[i]);
        tsum[i] += p;
        int mrow = i + ((lane >> 4) << 3);
        Pw[mrow * LKV + nj * 16 + ncol] = (bf16_t)p;
      }
#pragma unroll
    for (int i = 0; i < 8; ++i) {
      float t = tsum[i];
      t += __shfl_xor(t, 1, 32);
      t += __shfl_xor(t, 2, 32);
      t += __shfl_xor(t, 4, 32);
      t += __shfl_xor(t, 8, 32);
      rl[i] = rl[i] * corr[i] + t;
    }
#pragma unroll
    for (int nj = 0; nj < 8; ++nj)
#pragma unroll
      for (int i = 0; i < 8; ++i) ctx[nj][i] *= corr[i];

    // wave-private LDS RAW: DS ops are in-order per wave, make it explicit
    asm volatile("s_wait_dscnt 0" ::: "memory");

    // ctx += P @ V ; B-frag cols = head dim, K-dim = keys (from V^T tile)
#pragma unroll
    for (int kc = 0; kc < 4; ++kc) {
      v16bf pf = load_frag_a<136>(Pw + kc * 32);
#pragma unroll
      for (int nd = 0; nd < 8; ++nd) {
        v16bf vfr = load_frag_bt<136>(Vts + (nd * 16) * LKV + kc * 32);
        ctx[nd] = wmma_bf16(pf, vfr, ctx[nd]);
      }
    }
    __syncthreads();
  }

  // normalize and emit ctx in [B][S][H*128] for the output GEMM
#pragma unroll
  for (int nd = 0; nd < 8; ++nd)
#pragma unroll
    for (int i = 0; i < 8; ++i) {
      int s = qrow_base + i;
      int d = nd * 16 + ncol;
      float v = ctx[nd][i] / rl[i];
      CtxBuf[((size_t)(b * SEQ + s)) * D_MODEL + h * HDIM + d] = (bf16_t)v;
    }
}

// ---------------------------------------------------------------------------
// Kernel 3: output projection  out = ctx(bf16) @ Wo, using Wo^T bf16, f32 out.
// ---------------------------------------------------------------------------
__global__ __launch_bounds__(256)
void out_gemm_kernel(const bf16_t* __restrict__ A,
                     const bf16_t* __restrict__ WoT,   // [N=2048][K=2048] bf16
                     float* __restrict__ Out)
{
  const int n0 = blockIdx.x * 128;
  const int m0 = blockIdx.y * 128;
  const int tid = threadIdx.x, wave = tid >> 5, lane = tid & 31;

  __shared__ __attribute__((aligned(16))) bf16_t As[128][72];
  __shared__ __attribute__((aligned(16))) bf16_t Bs[128][72];

  const int wm = wave & 3, wn = wave >> 2;
  v8f acc[2][4] = {};

  for (int k0 = 0; k0 < D_MODEL; k0 += 64) {
    for (int i = tid; i < 128 * 16; i += 256) {
      int r = i >> 4, c4 = (i & 15) * 4;
      *(uint2*)&As[r][c4] = *(const uint2*)(A + (size_t)(m0 + r) * D_MODEL + k0 + c4);
    }
    for (int i = tid; i < 128 * 16; i += 256) {
      int r = i >> 4, c4 = (i & 15) * 4;
      *(uint2*)&Bs[r][c4] = *(const uint2*)(WoT + (size_t)(n0 + r) * D_MODEL + k0 + c4);
    }
    __syncthreads();

#pragma unroll
    for (int kc = 0; kc < 2; ++kc) {
      v16bf a0 = load_frag_a<72>(&As[wm * 32 + 0][kc * 32]);
      v16bf a1 = load_frag_a<72>(&As[wm * 32 + 16][kc * 32]);
#pragma unroll
      for (int nj = 0; nj < 4; ++nj) {
        v16bf bfr = load_frag_bt<72>(&Bs[wn * 64 + nj * 16][kc * 32]);
        acc[0][nj] = wmma_bf16(a0, bfr, acc[0][nj]);
        acc[1][nj] = wmma_bf16(a1, bfr, acc[1][nj]);
      }
    }
    __syncthreads();
  }

#pragma unroll
  for (int mi = 0; mi < 2; ++mi)
#pragma unroll
    for (int nj = 0; nj < 4; ++nj)
#pragma unroll
      for (int i = 0; i < 8; ++i) {
        int ml = wm * 32 + mi * 16 + i + ((lane >> 4) << 3);
        int nl = wn * 64 + nj * 16 + (lane & 15);
        Out[(size_t)(m0 + ml) * D_MODEL + n0 + nl] = acc[mi][nj][i];
      }
}

// ---------------------------------------------------------------------------
extern "C" void kernel_launch(void* const* d_in, const int* in_sizes, int n_in,
                              void* d_out, int out_size, void* d_ws, size_t ws_size,
                              hipStream_t stream) {
  (void)in_sizes; (void)n_in; (void)out_size; (void)ws_size;
  const float* X  = (const float*)d_in[0];
  const float* Wq = (const float*)d_in[1];
  const float* Wk = (const float*)d_in[2];
  const float* Wv = (const float*)d_in[3];
  const float* Wo = (const float*)d_in[4];
  float* Out = (float*)d_out;

  char* ws = (char*)d_ws;
  bf16_t* Qbuf = (bf16_t*)(ws);                  // 16 MB  [B][H][S][128]
  bf16_t* Kbuf = (bf16_t*)(ws + (16ull << 20));  //  4 MB  [B][Hkv][S][128]
  bf16_t* Vt   = (bf16_t*)(ws + (20ull << 20));  //  4 MB  [B][Hkv][128][S]
  bf16_t* Ctx  = (bf16_t*)(ws + (24ull << 20));  // 16 MB  [B][S][H*128]
  bf16_t* WT   = (bf16_t*)(ws + (40ull << 20));  // 12 MB  [3072][2048] = [Wq|Wk|Wv]^T
  bf16_t* WoT  = (bf16_t*)(ws + (52ull << 20));  //  8 MB  [2048][2048] = Wo^T

  // 0) one-shot weight transpose + bf16 convert
  transpose_cvt_kernel<<<dim3(64, 64), 256, 0, stream>>>(Wq, 2048, 2048, WT);
  transpose_cvt_kernel<<<dim3(16, 64), 256, 0, stream>>>(Wk, 2048, 512,
                                                         WT + (size_t)2048 * 2048);
  transpose_cvt_kernel<<<dim3(16, 64), 256, 0, stream>>>(Wv, 2048, 512,
                                                         WT + (size_t)2560 * 2048);
  transpose_cvt_kernel<<<dim3(64, 64), 256, 0, stream>>>(Wo, 2048, 2048, WoT);

  // 1) fused QKV projection: 4096 x 3072 output, K = 2048
  qkv_gemm_kernel<<<dim3(24, 32), 256, 0, stream>>>(X, WT, Qbuf, Kbuf, Vt);

  // 2) attention: 16 q-tiles x 16 heads x 2 batches; 102 KB dynamic LDS
  size_t smem = (size_t)(128 * 136 + 128 * 136 + 8 * 16 * 136) * sizeof(bf16_t);
  flash_attn_kernel<<<dim3(16, 16, 2), 256, smem, stream>>>(Qbuf, Kbuf, Vt, Ctx);

  // 3) output projection: 4096 x 2048, K = 2048
  out_gemm_kernel<<<dim3(16, 32), 256, 0, stream>>>(Ctx, WoT, Out);
}
